// AttentionModule_42116449305282
// MI455X (gfx1250) — compile-verified, matrix operand order checked by reading, not run on hardware
//
#include <hip/hip_runtime.h>
#include <hip/hip_bf16.h>

#define BQ   4
#define SEQ  2048
#define DIM  1024
#define MTOT (BQ * SEQ)   // 8192 flattened rows

typedef __bf16 bf16;
typedef __attribute__((ext_vector_type(16))) __bf16 v16bf;
typedef __attribute__((ext_vector_type(8)))  __bf16 v8bf;
typedef __attribute__((ext_vector_type(8)))  float  v8f;

// ---------------------------------------------------------------------------
// WMMA 16x16x32 bf16 fragment loaders (CDNA5 ISA 7.12.2 16-bit layouts):
//   lane&15  -> row within 16-row tile (A: M row; B: N column == row of B^T)
//   lane>>4  -> K phase: halves 0..7  = K + phase + {0..7}
//               halves 8..15 = K + phase + {16..23},  phase = (lane>>4)*8
// Caller passes a pointer already offset to (row, k0 + phase).
// ---------------------------------------------------------------------------
__device__ __forceinline__ v16bf load_frag(const bf16* __restrict__ p) {
    v16bf f;
    *(v8bf*)&f         = *(const v8bf*)(p);        // 16B-aligned (k mult. of 8)
    *(((v8bf*)&f) + 1) = *(const v8bf*)(p + 16);
    return f;
}

__device__ __forceinline__ v16bf load_frag(const float* __restrict__ p) {
    v16bf f;
#pragma unroll
    for (int i = 0; i < 8; ++i) {
        f[i]     = (__bf16)p[i];       // vectorizes to b128 loads + pk cvt
        f[i + 8] = (__bf16)p[i + 16];
    }
    return f;
}

// ---------------------------------------------------------------------------
// 64x64 wave macro-tile GEMM core (NT layout):
//   C[64x64] += A[64xK] * Bt[64xK]^T, fp32 accumulate via bf16 WMMA.
// 4x4 accumulators (128 VGPRs); 8 fragment loads feed 16 WMMAs per K-step —
// 2 WMMA per fragment load keeps the XDL pipe fed while VMEM is in flight.
// ---------------------------------------------------------------------------
template <typename TA, typename TB>
__device__ __forceinline__ void gemm_wave_64x64(const TA* __restrict__ A, size_t lda,
                                                const TB* __restrict__ Bt, size_t ldb,
                                                int K, int m0, int n0, int lane,
                                                v8f acc[4][4]) {
    const int r  = lane & 15;
    const int kp = (lane >> 4) << 3;   // K phase 0 or 8

    for (int k0 = 0; k0 < K; k0 += 32) {
        v16bf a[4], bm[4];
#pragma unroll
        for (int i = 0; i < 4; ++i) {
            const TA* pa = A + (size_t)(m0 + i * 16 + r) * lda + k0 + kp;
            __builtin_prefetch(pa + 32, 0, 1);
            a[i] = load_frag(pa);
        }
#pragma unroll
        for (int j = 0; j < 4; ++j) {
            const TB* pb = Bt + (size_t)(n0 + j * 16 + r) * ldb + k0 + kp;
            __builtin_prefetch(pb + 32, 0, 1);
            bm[j] = load_frag(pb);
        }
#pragma unroll
        for (int i = 0; i < 4; ++i)
#pragma unroll
            for (int j = 0; j < 4; ++j)
                acc[i][j] = __builtin_amdgcn_wmma_f32_16x16x32_bf16(
                    false, a[i], false, bm[j], (short)0, acc[i][j], false, false);
    }
}

// ---------------------------------------------------------------------------
// Projection: Out[m,n] = bf16( sum_d X[m,d]*W[n,d] + bias[n] )
//   X: [MTOT, DIM] fp32 row-major, W: [DIM, DIM] fp32 row-major  -> NT gemm.
//   TRANS=false: Out = [MTOT, DIM] bf16 row-major     (Q, K)
//   TRANS=true : Out = [B, DIM, SEQ] bf16 (V stored transposed per batch so
//                the PV gemm is also NT with contiguous K loads)
// Block = 128 thr = 4 waves (2x2); wave tile 64x64; block tile 128x128.
// ---------------------------------------------------------------------------
template <bool TRANS>
__global__ __launch_bounds__(128) void proj_kernel(const float* __restrict__ X,
                                                   const float* __restrict__ W,
                                                   const float* __restrict__ bias,
                                                   bf16* __restrict__ Out) {
    const int lane = threadIdx.x & 31;
    const int wv   = threadIdx.x >> 5;                    // 0..3
    const int m0   = blockIdx.y * 128 + (wv >> 1) * 64;
    const int n0   = blockIdx.x * 128 + (wv & 1) * 64;

    v8f acc[4][4] = {};
    gemm_wave_64x64(X, (size_t)DIM, W, (size_t)DIM, DIM, m0, n0, lane, acc);

    // C/D layout: element e of lane -> M = e + 8*(lane>>4), N = lane&15
    const int mb = (lane >> 4) << 3;
    const int nc = lane & 15;
#pragma unroll
    for (int i = 0; i < 4; ++i)
#pragma unroll
        for (int j = 0; j < 4; ++j) {
            const int   n  = n0 + j * 16 + nc;
            const float bv = bias[n];
#pragma unroll
            for (int e = 0; e < 8; ++e) {
                const int   m   = m0 + i * 16 + mb + e;
                const float val = acc[i][j][e] + bv;
                if (TRANS) {
                    const int bb   = m >> 11;         // m / SEQ
                    const int srow = m & (SEQ - 1);   // m % SEQ
                    Out[((size_t)bb * DIM + n) * SEQ + srow] = (bf16)val;
                } else {
                    Out[(size_t)m * DIM + n] = (bf16)val;
                }
            }
        }
}

// ---------------------------------------------------------------------------
// Scores: S[b,q,k] = (1/32) * sum_d Q[b,q,d]*K[b,k,d]   (NT, bf16 operands)
// ---------------------------------------------------------------------------
__global__ __launch_bounds__(128) void scores_kernel(const bf16* __restrict__ Q,
                                                     const bf16* __restrict__ K,
                                                     float* __restrict__ Scr) {
    const int b = blockIdx.z;
    const bf16* Qb = Q + (size_t)b * SEQ * DIM;
    const bf16* Kb = K + (size_t)b * SEQ * DIM;
    float*      Sb = Scr + (size_t)b * SEQ * SEQ;

    const int lane = threadIdx.x & 31;
    const int wv   = threadIdx.x >> 5;
    const int m0   = blockIdx.y * 128 + (wv >> 1) * 64;
    const int n0   = blockIdx.x * 128 + (wv & 1) * 64;

    v8f acc[4][4] = {};
    gemm_wave_64x64(Qb, (size_t)DIM, Kb, (size_t)DIM, DIM, m0, n0, lane, acc);

    const int mb = (lane >> 4) << 3;
    const int nc = lane & 15;
    const float scale = 0.03125f;  // 1/sqrt(1024)
#pragma unroll
    for (int i = 0; i < 4; ++i)
#pragma unroll
        for (int j = 0; j < 4; ++j)
#pragma unroll
            for (int e = 0; e < 8; ++e) {
                const int m = m0 + i * 16 + mb + e;
                const int n = n0 + j * 16 + nc;
                Sb[(size_t)m * SEQ + n] = acc[i][j][e] * scale;
            }
}

// ---------------------------------------------------------------------------
// Row softmax over SEQ=2048, in place (this buffer IS the `weights` output).
// One 256-thread block per row, 8 elements/thread, LDS tree reductions.
// ---------------------------------------------------------------------------
__global__ __launch_bounds__(256) void softmax_kernel(float* __restrict__ Scr) {
    __shared__ float red[256];
    float* row = Scr + (size_t)blockIdx.x * SEQ;
    const int t = threadIdx.x;

    float vals[8];
    float mx = -INFINITY;
#pragma unroll
    for (int i = 0; i < 8; ++i) {
        vals[i] = row[t + i * 256];
        mx = fmaxf(mx, vals[i]);
    }
    red[t] = mx;
    __syncthreads();
    for (int s = 128; s > 0; s >>= 1) {
        if (t < s) red[t] = fmaxf(red[t], red[t + s]);
        __syncthreads();
    }
    mx = red[0];
    __syncthreads();

    float sum = 0.f;
#pragma unroll
    for (int i = 0; i < 8; ++i) {
        vals[i] = __expf(vals[i] - mx);
        sum += vals[i];
    }
    red[t] = sum;
    __syncthreads();
    for (int s = 128; s > 0; s >>= 1) {
        if (t < s) red[t] += red[t + s];
        __syncthreads();
    }
    const float inv = 1.0f / red[0];
#pragma unroll
    for (int i = 0; i < 8; ++i) row[t + i * 256] = vals[i] * inv;
}

// ---------------------------------------------------------------------------
// Output: O[b,q,d] = sum_k P[b,q,k] * V[b,k,d]
//   P fp32 (converted to bf16 in-register), Vt bf16 stored [B, DIM, SEQ] -> NT.
// ---------------------------------------------------------------------------
__global__ __launch_bounds__(128) void av_kernel(const float* __restrict__ P,
                                                 const bf16* __restrict__ Vt,
                                                 float* __restrict__ O) {
    const int b = blockIdx.z;
    const float* Pb = P + (size_t)b * SEQ * SEQ;   // [SEQ, SEQ]
    const bf16*  Vb = Vt + (size_t)b * DIM * SEQ;  // [DIM, SEQ]
    float*       Ob = O + (size_t)b * SEQ * DIM;   // [SEQ, DIM]

    const int lane = threadIdx.x & 31;
    const int wv   = threadIdx.x >> 5;
    const int m0   = blockIdx.y * 128 + (wv >> 1) * 64;   // query rows
    const int n0   = blockIdx.x * 128 + (wv & 1) * 64;    // d columns

    v8f acc[4][4] = {};
    gemm_wave_64x64(Pb, (size_t)SEQ, Vb, (size_t)SEQ, SEQ, m0, n0, lane, acc);

    const int mb = (lane >> 4) << 3;
    const int nc = lane & 15;
#pragma unroll
    for (int i = 0; i < 4; ++i)
#pragma unroll
        for (int j = 0; j < 4; ++j)
#pragma unroll
            for (int e = 0; e < 8; ++e) {
                const int m = m0 + i * 16 + mb + e;
                const int n = n0 + j * 16 + nc;
                Ob[(size_t)m * DIM + n] = acc[i][j][e];
            }
}

// ---------------------------------------------------------------------------
extern "C" void kernel_launch(void* const* d_in, const int* in_sizes, int n_in,
                              void* d_out, int out_size, void* d_ws, size_t ws_size,
                              hipStream_t stream) {
    const float* query = (const float*)d_in[0];
    const float* key   = (const float*)d_in[1];
    const float* value = (const float*)d_in[2];
    const float* Wq    = (const float*)d_in[3];
    const float* bq    = (const float*)d_in[4];
    const float* Wk    = (const float*)d_in[5];
    const float* bk    = (const float*)d_in[6];
    const float* Wv    = (const float*)d_in[7];
    const float* bv    = (const float*)d_in[8];

    float* out = (float*)d_out;                        // [B,S,D]  attention_output
    float* wts = out + (size_t)MTOT * DIM;             // [B,S,S]  weights (in place)

    bf16* Qb = (bf16*)d_ws;                            // [MTOT, DIM]
    bf16* Kb = Qb + (size_t)MTOT * DIM;                // [MTOT, DIM]
    bf16* Vt = Kb + (size_t)MTOT * DIM;                // [B, DIM, SEQ]

    const dim3 blk(128);
    proj_kernel<false><<<dim3(DIM / 128, MTOT / 128), blk, 0, stream>>>(query, Wq, bq, Qb);
    proj_kernel<false><<<dim3(DIM / 128, MTOT / 128), blk, 0, stream>>>(key,   Wk, bk, Kb);
    proj_kernel<true ><<<dim3(DIM / 128, MTOT / 128), blk, 0, stream>>>(value, Wv, bv, Vt);

    scores_kernel<<<dim3(SEQ / 128, SEQ / 128, BQ), blk, 0, stream>>>(Qb, Kb, wts);
    softmax_kernel<<<dim3(MTOT), 256, 0, stream>>>(wts);
    av_kernel<<<dim3(DIM / 128, SEQ / 128, BQ), blk, 0, stream>>>(wts, Vt, out);
}
